// MyDictionary_74414603370755
// MI455X (gfx1250) — matrix operand order repaired and checked by reading, not compiled, not run on hardware
//
#include <hip/hip_runtime.h>
#include <math.h>

// ---------------------------------------------------------------------------
// Types for CDNA5 WMMA (gfx1250, wave32)
// ---------------------------------------------------------------------------
typedef __bf16 v16bf __attribute__((ext_vector_type(16)));
typedef float  v8f   __attribute__((ext_vector_type(8)));

__device__ __forceinline__ unsigned short f32_to_bf16(float f) {
  unsigned int u = __float_as_uint(f);
  unsigned int r = u + 0x7FFFu + ((u >> 16) & 1u);   // round-to-nearest-even
  return (unsigned short)(r >> 16);
}

// Async 16B global -> LDS copy (CDNA5, ASYNCcnt-tracked, no VGPR staging)
__device__ __forceinline__ void async_g2l_b128(unsigned lds_off,
                                               const void* gaddr) {
  asm volatile("global_load_async_to_lds_b128 %0, %1, off"
               :: "v"(lds_off), "v"((unsigned long long)gaddr) : "memory");
}
__device__ __forceinline__ void wait_async0() {
  asm volatile("s_wait_asynccnt 0" ::: "memory");
}

// ---------------------------------------------------------------------------
// Tiled bf16 GEMM with WMMA:  out = alpha * A(bf16) @ B(bf16) [+bias] [+resid]
//   A: [M,K] row-major (lda), batched via strideA (elements)
//   B: transB==0 -> [K,N] row-major (ldb); transB==1 -> [N,K] row-major (A@B^T)
//   out f32 (optional) and/or outb bf16 (optional), same ldo/strideO
// Block: 256 threads = 8 waves (4x2); block tile 128x128; wave tile 32x64
// ---------------------------------------------------------------------------
#define TBM 128
#define TBN 128
#define TKT 32
#define LDSK 40   // padded K stride (bf16 elements) -> 80B rows, 16B aligned

__global__ __launch_bounds__(256) void gemm_bf16_kernel(
    const unsigned short* __restrict__ A, int lda, long long strideA,
    const unsigned short* __restrict__ Bm, int ldb, long long strideB, int transB,
    const float* __restrict__ bias,
    const float* __restrict__ resid, int ldr, long long strideR,
    float* __restrict__ outF, unsigned short* __restrict__ outB,
    int ldo, long long strideO,
    float alpha, int M, int N, int K)
{
  __shared__ unsigned short As[TBM][LDSK];
  __shared__ unsigned short Bs[TBN][LDSK];

  const int tid  = threadIdx.x;
  const int wave = tid >> 5;
  const int lane = tid & 31;
  const int waveM = wave >> 1;   // 0..3 -> 32-row slices
  const int waveN = wave & 1;    // 0..1 -> 64-col slices
  const int hlf   = lane >> 4;   // half-wave select
  const int lm    = lane & 15;

  const int z = blockIdx.z;
  const unsigned short* Ab = A  + (long long)z * strideA;
  const unsigned short* Bb = Bm + (long long)z * strideB;

  const int m_block = blockIdx.y * TBM;
  const int n_block = blockIdx.x * TBN;

  const bool fullA = (m_block + TBM) <= M;
  const bool fullB = (n_block + TBN) <= N;

  const v8f zero8 = {0.f,0.f,0.f,0.f,0.f,0.f,0.f,0.f};
  v8f acc[2][4];
  #pragma unroll
  for (int i = 0; i < 2; i++)
    #pragma unroll
    for (int j = 0; j < 4; j++) acc[i][j] = zero8;

  for (int k0 = 0; k0 < K; k0 += TKT) {
    const bool fullK = (k0 + TKT) <= K;
    bool used_async = false;

    // ---- load A tile (128x32): two 16B chunks per thread ----
    if (fullA && fullK) {
      #pragma unroll
      for (int i = 0; i < 2; i++) {
        int c  = tid + i * 256;            // 0..511
        int r  = c >> 2;                   // 0..127
        int kc = (c & 3) * 8;              // 0,8,16,24
        async_g2l_b128((unsigned)(unsigned long long)&As[r][kc],
                       Ab + (long long)(m_block + r) * lda + (k0 + kc));
      }
      used_async = true;
    } else {
      #pragma unroll
      for (int i = 0; i < 2; i++) {
        int c  = tid + i * 256;
        int r  = c >> 2;
        int kc = (c & 3) * 8;
        int gm = m_block + r;
        int gk = k0 + kc;
        uint4 val = {0u,0u,0u,0u};
        if (gm < M && (gk + 8) <= K)
          val = *(const uint4*)(Ab + (long long)gm * lda + gk);
        *(uint4*)&As[r][kc] = val;
      }
    }

    // ---- load B tile (store as Bs[n][k], k-contiguous) ----
    if (transB) {
      if (fullB && fullK) {
        #pragma unroll
        for (int i = 0; i < 2; i++) {
          int c  = tid + i * 256;
          int n  = c >> 2;
          int kc = (c & 3) * 8;
          async_g2l_b128((unsigned)(unsigned long long)&Bs[n][kc],
                         Bb + (long long)(n_block + n) * ldb + (k0 + kc));
        }
        used_async = true;
      } else {
        #pragma unroll
        for (int i = 0; i < 2; i++) {
          int c  = tid + i * 256;
          int n  = c >> 2;
          int kc = (c & 3) * 8;
          int gn = n_block + n, gk = k0 + kc;
          uint4 val = {0u,0u,0u,0u};
          if (gn < N && (gk + 8) <= K)
            val = *(const uint4*)(Bb + (long long)gn * ldb + gk);
          *(uint4*)&Bs[n][kc] = val;
        }
      }
    } else {
      // B[k][n]: read contiguous in n, scatter-transpose into LDS
      #pragma unroll
      for (int i = 0; i < 2; i++) {
        int c  = tid + i * 256;      // 0..511
        int kk = c >> 4;             // 0..31
        int n0 = (c & 15) * 8;
        int gk = k0 + kk;
        int gnb = n_block + n0;
        unsigned short tmp[8] = {0,0,0,0,0,0,0,0};
        if (gk < K) {
          if (gnb + 8 <= N) {
            uint4 val = *(const uint4*)(Bb + (long long)gk * ldb + gnb);
            const unsigned short* pv = (const unsigned short*)&val;
            #pragma unroll
            for (int j = 0; j < 8; j++) tmp[j] = pv[j];
          } else {
            for (int j = 0; j < 8; j++)
              if (gnb + j < N) tmp[j] = Bb[(long long)gk * ldb + gnb + j];
          }
        }
        #pragma unroll
        for (int j = 0; j < 8; j++) Bs[n0 + j][kk] = tmp[j];
      }
    }
    if (used_async) wait_async0();   // drain this wave's async copies
    __syncthreads();

    // ---- build fragments per ISA VGPR layouts ----
    union Frag { unsigned int u[8]; v16bf v; };
    v16bf afr[2], bfr[4];
    #pragma unroll
    for (int tm = 0; tm < 2; tm++) {
      int m = waveM * 32 + tm * 16 + lm;
      Frag fa;
      #pragma unroll
      for (int vv = 0; vv < 8; vv++) {
        int koff = (vv < 4) ? (2 * vv + 8 * hlf) : (16 + 2 * (vv - 4) + 8 * hlf);
        fa.u[vv] = *(const unsigned int*)&As[m][koff];
      }
      afr[tm] = fa.v;
    }
    #pragma unroll
    for (int tn = 0; tn < 4; tn++) {
      int n = waveN * 64 + tn * 16 + lm;
      Frag fb;
      #pragma unroll
      for (int vv = 0; vv < 8; vv++) {
        int koff = (vv < 4) ? (2 * vv + 8 * hlf) : (16 + 2 * (vv - 4) + 8 * hlf);
        fb.u[vv] = *(const unsigned int*)&Bs[n][koff];
      }
      bfr[tn] = fb.v;
    }

    #pragma unroll
    for (int tm = 0; tm < 2; tm++)
      #pragma unroll
      for (int tn = 0; tn < 4; tn++)
        acc[tm][tn] = __builtin_amdgcn_wmma_f32_16x16x32_bf16(
            false, afr[tm], false, bfr[tn], (short)0, acc[tm][tn], false, false);

    __syncthreads();
  }

  // ---- epilogue: C/D layout -> row = r + 8*half, col = lane&15 ----
  #pragma unroll
  for (int tm = 0; tm < 2; tm++) {
    #pragma unroll
    for (int tn = 0; tn < 4; tn++) {
      #pragma unroll
      for (int r = 0; r < 8; r++) {
        int gm = m_block + waveM * 32 + tm * 16 + r + 8 * hlf;
        int gn = n_block + waveN * 64 + tn * 16 + lm;
        if (gm < M && gn < N) {
          float v = alpha * acc[tm][tn][r];
          if (bias)  v += bias[gn];
          if (resid) v += resid[(long long)z * strideR + (long long)gm * ldr + gn];
          long long o = (long long)z * strideO + (long long)gm * ldo + gn;
          if (outF) outF[o] = v;
          if (outB) outB[o] = f32_to_bf16(v);
        }
      }
    }
  }
}

// ---------------------------------------------------------------------------
// LayerNorm over last dim (C); grid = rows. Outputs f32 and/or bf16.
// ---------------------------------------------------------------------------
__global__ __launch_bounds__(256) void ln_kernel(
    const float* __restrict__ x, const float* __restrict__ w,
    const float* __restrict__ b, float* __restrict__ outf,
    unsigned short* __restrict__ outb, int Cn)
{
  __shared__ float red[256];
  const int tid = threadIdx.x;
  const long long row = blockIdx.x;
  const float* xr = x + row * (long long)Cn;

  float s = 0.f;
  for (int c = tid; c < Cn; c += 256) s += xr[c];
  red[tid] = s; __syncthreads();
  for (int o = 128; o > 0; o >>= 1) { if (tid < o) red[tid] += red[tid + o]; __syncthreads(); }
  float mean = red[0] / (float)Cn; __syncthreads();

  float s2 = 0.f;
  for (int c = tid; c < Cn; c += 256) { float d = xr[c] - mean; s2 += d * d; }
  red[tid] = s2; __syncthreads();
  for (int o = 128; o > 0; o >>= 1) { if (tid < o) red[tid] += red[tid + o]; __syncthreads(); }
  float rstd = rsqrtf(red[0] / (float)Cn + 1e-5f);

  for (int c = tid; c < Cn; c += 256) {
    float y = (xr[c] - mean) * rstd * w[c] + b[c];
    if (outf) outf[row * (long long)Cn + c] = y;
    if (outb) outb[row * (long long)Cn + c] = f32_to_bf16(y);
  }
}

// ---------------------------------------------------------------------------
// Row softmax f32 -> bf16; grid = rows, row length Nn
// ---------------------------------------------------------------------------
__global__ __launch_bounds__(256) void softmax_kernel(
    const float* __restrict__ in, unsigned short* __restrict__ outb, int Nn)
{
  __shared__ float red[256];
  const int tid = threadIdx.x;
  const long long row = blockIdx.x;
  const float* inr = in + row * (long long)Nn;

  float mx = -3.4e38f;
  for (int i = tid; i < Nn; i += 256) mx = fmaxf(mx, inr[i]);
  red[tid] = mx; __syncthreads();
  for (int o = 128; o > 0; o >>= 1) { if (tid < o) red[tid] = fmaxf(red[tid], red[tid + o]); __syncthreads(); }
  mx = red[0]; __syncthreads();

  float s = 0.f;
  for (int i = tid; i < Nn; i += 256) s += __expf(inr[i] - mx);
  red[tid] = s; __syncthreads();
  for (int o = 128; o > 0; o >>= 1) { if (tid < o) red[tid] += red[tid + o]; __syncthreads(); }
  float inv = 1.f / red[0];

  unsigned short* outr = outb + row * (long long)Nn;
  for (int i = tid; i < Nn; i += 256) outr[i] = f32_to_bf16(__expf(inr[i] - mx) * inv);
}

// ---------------------------------------------------------------------------
// Sparsemax per row (Nn must be 1024): bitonic sort + scan + threshold
// ---------------------------------------------------------------------------
__global__ __launch_bounds__(256) void sparsemax_kernel(
    const float* __restrict__ in, unsigned short* __restrict__ outb, int Nn)
{
  __shared__ float zb[1024];
  __shared__ float sb[1024];
  __shared__ float psum[256];
  __shared__ int   icnt[256];
  __shared__ float tau_s;
  const int tid = threadIdx.x;
  const long long row = blockIdx.x;
  const float* inr = in + row * (long long)Nn;

  for (int i = tid; i < Nn; i += 256) { float v = inr[i]; zb[i] = v; sb[i] = v; }
  __syncthreads();

  // bitonic sort descending
  for (int kk = 2; kk <= Nn; kk <<= 1) {
    for (int j = kk >> 1; j > 0; j >>= 1) {
      for (int i = tid; i < Nn; i += 256) {
        int ixj = i ^ j;
        if (ixj > i) {
          float a = sb[i], b = sb[ixj];
          bool desc = ((i & kk) == 0);
          if (desc ? (a < b) : (a > b)) { sb[i] = b; sb[ixj] = a; }
        }
      }
      __syncthreads();
    }
  }

  // inclusive cumsum: 4 elements per thread + block scan of partials
  float loc[4]; float run = 0.f;
  const int base = tid * 4;
  #pragma unroll
  for (int q = 0; q < 4; q++) { loc[q] = sb[base + q]; run += loc[q]; }
  psum[tid] = run; __syncthreads();
  for (int o = 1; o < 256; o <<= 1) {
    float v = (tid >= o) ? psum[tid - o] : 0.f;
    __syncthreads();
    psum[tid] += v;
    __syncthreads();
  }
  float c = psum[tid] - run;  // exclusive prefix
  int cloc = 0; float cums[4];
  #pragma unroll
  for (int q = 0; q < 4; q++) {
    c += loc[q];
    cums[q] = c;
    if ((float)(base + q + 1) * loc[q] > c - 1.f) cloc++;
  }
  __syncthreads();
  #pragma unroll
  for (int q = 0; q < 4; q++) sb[base + q] = cums[q];
  icnt[tid] = cloc; __syncthreads();
  for (int o = 128; o > 0; o >>= 1) { if (tid < o) icnt[tid] += icnt[tid + o]; __syncthreads(); }
  if (tid == 0) {
    int kk = icnt[0]; if (kk < 1) kk = 1;
    tau_s = (sb[kk - 1] - 1.f) / (float)kk;
  }
  __syncthreads();
  float tau = tau_s;
  unsigned short* outr = outb + row * (long long)Nn;
  for (int i = tid; i < Nn; i += 256) {
    float v = zb[i] - tau; if (v < 0.f) v = 0.f;
    outr[i] = f32_to_bf16(v);
  }
}

// ---------------------------------------------------------------------------
// 3x3 same-pad average pool on [B, H*H, C] layout (pad with zeros, /9)
// ---------------------------------------------------------------------------
__global__ __launch_bounds__(256) void pool3_kernel(
    const float* __restrict__ in, float* __restrict__ out, int Bn, int Hn, int Cn)
{
  long long total = (long long)Bn * Hn * Hn * Cn;
  long long idx = (long long)blockIdx.x * 256 + threadIdx.x;
  if (idx >= total) return;
  int c = (int)(idx % Cn);
  long long t = idx / Cn;
  int l = (int)(t % (Hn * Hn));
  long long b = t / (Hn * Hn);
  int y = l / Hn, x = l % Hn;
  float s = 0.f;
  for (int dy = -1; dy <= 1; dy++)
    for (int dx = -1; dx <= 1; dx++) {
      int yy = y + dy, xx = x + dx;
      if (yy >= 0 && yy < Hn && xx >= 0 && xx < Hn)
        s += in[(b * Hn * Hn + (long long)yy * Hn + xx) * Cn + c];
    }
  out[idx] = s / 9.f;
}

// ---------------------------------------------------------------------------
// Per-row cosine distance: dis[row] = 1 - cos(a[row,:], r[row,:])
// ---------------------------------------------------------------------------
__global__ __launch_bounds__(256) void dis_kernel(
    const float* __restrict__ a, const float* __restrict__ r,
    float* __restrict__ dis, int Cn)
{
  __shared__ float r0[256], r1[256], r2[256];
  const int tid = threadIdx.x;
  const long long row = blockIdx.x;
  const float* ar = a + row * (long long)Cn;
  const float* rr = r + row * (long long)Cn;
  float sab = 0.f, saa = 0.f, sbb = 0.f;
  for (int c = tid; c < Cn; c += 256) {
    float x = ar[c], y = rr[c];
    sab += x * y; saa += x * x; sbb += y * y;
  }
  r0[tid] = sab; r1[tid] = saa; r2[tid] = sbb; __syncthreads();
  for (int o = 128; o > 0; o >>= 1) {
    if (tid < o) { r0[tid] += r0[tid + o]; r1[tid] += r1[tid + o]; r2[tid] += r2[tid + o]; }
    __syncthreads();
  }
  if (tid == 0) {
    float na = fmaxf(sqrtf(r1[0]), 1e-8f);
    float nb = fmaxf(sqrtf(r2[0]), 1e-8f);
    dis[row] = 1.f - r0[0] / (na * nb);
  }
}

// ---------------------------------------------------------------------------
// Masked loss reductions; accumulates into out2[0]=lossC, out2[1]=lossQ
// ---------------------------------------------------------------------------
__global__ __launch_bounds__(256) void loss_kernel(
    const float* __restrict__ dis, const float* __restrict__ gtn,
    const float* __restrict__ gta, float* __restrict__ out2, int Bn, int Ln)
{
  __shared__ float red[256];
  __shared__ float lnv[8], lav[8];
  const int tid = threadIdx.x;
  for (int b = 0; b < Bn; b++) {
    float sdn = 0.f, sn = 0.f, sda = 0.f, sa = 0.f;
    for (int l = tid; l < Ln; l += 256) {
      float d = dis[(long long)b * Ln + l];
      float n = gtn[(long long)b * Ln + l];
      float a = gta[(long long)b * Ln + l];
      sdn += d * n; sn += n; sda += d * a; sa += a;
    }
    float vals[4] = { sdn, sn, sda, sa };
    float outv[4];
    for (int q = 0; q < 4; q++) {
      red[tid] = vals[q]; __syncthreads();
      for (int o = 128; o > 0; o >>= 1) { if (tid < o) red[tid] += red[tid + o]; __syncthreads(); }
      outv[q] = red[0]; __syncthreads();
    }
    if (tid == 0) {
      lnv[b] = outv[0] / (outv[1] + 1e-6f);
      lav[b] = outv[2] / (outv[3] + 1e-6f);
    }
    __syncthreads();
  }
  if (tid == 0) {
    float st = 0.f, sln = 0.f; int cnt = 0;
    for (int b = 0; b < Bn; b++) {
      float t = lnv[b] - lav[b];
      if (lav[b] == 0.f) t = 0.f;
      if (t < 0.f) t = 0.f;
      st += t; if (t != 0.f) cnt++;
      sln += lnv[b];
    }
    out2[0] += st / ((float)cnt + 1e-6f);
    out2[1] += sln / (float)Bn;
  }
}

__global__ void cvt_bf16_kernel(const float* __restrict__ in,
                                unsigned short* __restrict__ out, long long n) {
  long long i = (long long)blockIdx.x * 256 + threadIdx.x;
  if (i < n) out[i] = f32_to_bf16(in[i]);
}

__global__ void zero2_kernel(float* out) { out[0] = 0.f; out[1] = 0.f; }

// ---------------------------------------------------------------------------
// Host orchestration
// ---------------------------------------------------------------------------
extern "C" void kernel_launch(void* const* d_in, const int* in_sizes, int n_in,
                              void* d_out, int out_size, void* d_ws, size_t ws_size,
                              hipStream_t stream) {
  (void)in_sizes; (void)n_in; (void)out_size; (void)ws_size;

  const int Bn = 4, Ls = 1024, Cc = 768, HD = 192, Hh = 32;
  const int BL = Bn * Ls;
  const long long BLC = (long long)Bn * Ls * Cc;
  const long long BLL = (long long)Bn * Ls * Ls;

  const float* ano  = (const float*)d_in[0];
  const float* good = (const float*)d_in[1];
  const float* gtn  = (const float*)d_in[2];
  const float* gta  = (const float*)d_in[3];

  // jax pytree flatten: outer dict keys sorted ('k','q','v'); inner keys sorted:
  // l1b,l1w,l2b,l2w,l3w,n1b,n1w,nb,nw,wk,wq,wv
  struct BlkW {
    const float *l1b,*l1w,*l2b,*l2w,*l3w,*n1b,*n1w,*nb,*nw,*wk,*wq,*wv;
    unsigned short *l1w_b,*l2w_b,*l3w_b,*wk_b,*wq_b,*wv_b;
  };
  BlkW blk[3];
  for (int ib = 0; ib < 3; ib++) {
    const float* P[12];
    for (int j = 0; j < 12; j++) P[j] = (const float*)d_in[4 + ib * 12 + j];
    blk[ib].l1b = P[0];  blk[ib].l1w = P[1];  blk[ib].l2b = P[2]; blk[ib].l2w = P[3];
    blk[ib].l3w = P[4];  blk[ib].n1b = P[5];  blk[ib].n1w = P[6]; blk[ib].nb  = P[7];
    blk[ib].nw  = P[8];  blk[ib].wk  = P[9];  blk[ib].wq  = P[10]; blk[ib].wv = P[11];
  }

  // workspace allocator
  char* wp = (char*)d_ws;
  auto alloc = [&](size_t bytes) -> void* {
    void* p = (void*)wp; wp += (bytes + 255) & ~(size_t)255; return p;
  };

  // bf16 weights (converted every call; deterministic)
  const long long WN = (long long)Cc * Cc;
  for (int ib = 0; ib < 3; ib++) {
    blk[ib].l1w_b = (unsigned short*)alloc(WN * 2);
    blk[ib].l2w_b = (unsigned short*)alloc(WN * 2);
    blk[ib].l3w_b = (unsigned short*)alloc(WN * 2);
    blk[ib].wk_b  = (unsigned short*)alloc(WN * 2);
    blk[ib].wq_b  = (unsigned short*)alloc(WN * 2);
    blk[ib].wv_b  = (unsigned short*)alloc(WN * 2);
    int g = (int)((WN + 255) / 256);
    cvt_bf16_kernel<<<g, 256, 0, stream>>>(blk[ib].l1w, blk[ib].l1w_b, WN);
    cvt_bf16_kernel<<<g, 256, 0, stream>>>(blk[ib].l2w, blk[ib].l2w_b, WN);
    cvt_bf16_kernel<<<g, 256, 0, stream>>>(blk[ib].l3w, blk[ib].l3w_b, WN);
    cvt_bf16_kernel<<<g, 256, 0, stream>>>(blk[ib].wk,  blk[ib].wk_b,  WN);
    cvt_bf16_kernel<<<g, 256, 0, stream>>>(blk[ib].wq,  blk[ib].wq_b,  WN);
    cvt_bf16_kernel<<<g, 256, 0, stream>>>(blk[ib].wv,  blk[ib].wv_b,  WN);
  }

  // pipeline buffers
  float* qf_p = (float*)alloc(BLC * 4);
  float* sf_p = (float*)alloc(BLC * 4);
  unsigned short* FQb = (unsigned short*)alloc(BLC * 2);
  unsigned short* FKb = (unsigned short*)alloc(BLC * 2);
  unsigned short* FVb = (unsigned short*)alloc(BLC * 2);
  float*          attnF = (float*)alloc(BLL * 4);
  unsigned short* attnB = (unsigned short*)alloc(BLL * 2);
  float* Rw   = (float*)alloc(BLC * 4);
  float* disb = (float*)alloc((long long)BL * 4);
  // attn-block scratch (reused for all 6 invocations)
  float*          t_h   = (float*)alloc(BLC * 4);
  unsigned short* t_hb  = (unsigned short*)alloc(BLC * 2);
  unsigned short* t_qb  = (unsigned short*)alloc(BLC * 2);
  unsigned short* t_kb  = (unsigned short*)alloc(BLC * 2);
  unsigned short* t_vb  = (unsigned short*)alloc(BLC * 2);
  unsigned short* t_ob  = (unsigned short*)alloc(BLC * 2);
  unsigned short* t_x1b = (unsigned short*)alloc(BLC * 2);
  float*          t_x2  = (float*)alloc(BLC * 4);
  unsigned short* t_lnb = (unsigned short*)alloc(BLC * 2);

  auto gemm = [&](const unsigned short* A, int lda, long long sA,
                  const unsigned short* Bm, int ldb, long long sB, int transB,
                  const float* bias, const float* resid, int ldr, long long sR,
                  float* oF, unsigned short* oB, int ldo, long long sO,
                  float alpha, int M, int N, int K, int Z) {
    dim3 g((N + TBN - 1) / TBN, (M + TBM - 1) / TBM, Z);
    gemm_bf16_kernel<<<g, 256, 0, stream>>>(A, lda, sA, Bm, ldb, sB, transB,
                                            bias, resid, ldr, sR,
                                            oF, oB, ldo, sO, alpha, M, N, K);
  };

  const float iscaleH = 1.0f / sqrtf((float)HD);
  const float iscaleC = 1.0f / sqrtf((float)Cc);
  const long long sLC = (long long)Ls * Cc;
  const long long sLL = (long long)Ls * Ls;

  // Attn_Block: x -> (outB bf16 [, outF f32]), optional residual on final gemm
  auto run_block = [&](const BlkW& p, const float* x,
                       unsigned short* outB, float* outF, const float* resid) {
    ln_kernel<<<BL, 256, 0, stream>>>(x, p.n1w, p.n1b, t_h, t_hb, Cc);
    gemm(t_hb, Cc, 0, p.wq_b, Cc, 0, 0, nullptr, nullptr, 0, 0,
         nullptr, t_qb, Cc, 0, 1.f, BL, Cc, Cc, 1);
    gemm(t_hb, Cc, 0, p.wk_b, Cc, 0, 0, nullptr, nullptr, 0, 0,
         nullptr, t_kb, Cc, 0, 1.f, BL, Cc, Cc, 1);
    gemm(t_hb, Cc, 0, p.wv_b, Cc, 0, 0, nullptr, nullptr, 0, 0,
         nullptr, t_vb, Cc, 0, 1.f, BL, Cc, Cc, 1);
    for (int h = 0; h < 4; h++) {
      // scores = (q @ k^T) / sqrt(hd), batched over B
      gemm(t_qb + h * HD, Cc, sLC, t_kb + h * HD, Cc, sLC, 1,
           nullptr, nullptr, 0, 0, attnF, nullptr, Ls, sLL,
           iscaleH, Ls, Ls, HD, Bn);
      softmax_kernel<<<BL, 256, 0, stream>>>(attnF, attnB, Ls);
      // o = softmax(scores) @ v
      gemm(attnB, Ls, sLL, t_vb + h * HD, Cc, sLC, 0,
           nullptr, nullptr, 0, 0, nullptr, t_ob + h * HD, Cc, sLC,
           1.f, Ls, HD, Ls, Bn);
    }
    // x1 = o @ l1w + l1b + h(residual)
    gemm(t_ob, Cc, 0, p.l1w_b, Cc, 0, 0, p.l1b, t_h, Cc, 0,
         nullptr, t_x1b, Cc, 0, 1.f, BL, Cc, Cc, 1);
    // x2 = x1 @ l2w + l2b
    gemm(t_x1b, Cc, 0, p.l2w_b, Cc, 0, 0, p.l2b, nullptr, 0, 0,
         t_x2, nullptr, Cc, 0, 1.f, BL, Cc, Cc, 1);
    ln_kernel<<<BL, 256, 0, stream>>>(t_x2, p.nw, p.nb, nullptr, t_lnb, Cc);
    // out = LN(x2) @ l3w [+ resid]
    gemm(t_lnb, Cc, 0, p.l3w_b, Cc, 0, 0, nullptr, resid, Cc, 0,
         outF, outB, Cc, 0, 1.f, BL, Cc, Cc, 1);
  };

  zero2_kernel<<<1, 1, 0, stream>>>((float*)d_out);

  for (int scale = 0; scale < 2; scale++) {
    const float *qf, *sf;
    if (scale == 0) { qf = ano; sf = good; }
    else {
      int g = (int)((BLC + 255) / 256);
      pool3_kernel<<<g, 256, 0, stream>>>(ano,  qf_p, Bn, Hh, Cc);
      pool3_kernel<<<g, 256, 0, stream>>>(good, sf_p, Bn, Hh, Cc);
      qf = qf_p; sf = sf_p;
    }
    run_block(blk[1], qf, FQb, nullptr, nullptr);      // FQ (params 'q')
    run_block(blk[0], sf, FKb, nullptr, nullptr);      // FK (params 'k')
    run_block(blk[2], sf, FVb, nullptr, sf);           // FV = sf + block(sf)

    // attn = sparsemax( FQ @ FK^T * C^-0.5 )
    gemm(FQb, Cc, sLC, FKb, Cc, sLC, 1, nullptr, nullptr, 0, 0,
         attnF, nullptr, Ls, sLL, iscaleC, Ls, Ls, Cc, Bn);
    sparsemax_kernel<<<BL, 256, 0, stream>>>(attnF, attnB, Ls);

    // R = attn @ FV  (ks==1 writes retr straight into d_out+2)
    float* Rout = (scale == 0) ? ((float*)d_out + 2) : Rw;
    gemm(attnB, Ls, sLL, FVb, Cc, sLC, 0, nullptr, nullptr, 0, 0,
         Rout, nullptr, Cc, sLC, 1.f, Ls, Cc, Ls, Bn);

    dis_kernel<<<BL, 256, 0, stream>>>(qf, Rout, disb, Cc);
    loss_kernel<<<1, 256, 0, stream>>>(disb, gtn, gta, (float*)d_out, Bn, Ls);
  }
}
